// GraphNetBlock_4947802325261
// MI455X (gfx1250) — matrix operand order repaired and checked by reading, not compiled, not run on hardware
//
#include <hip/hip_runtime.h>
#include <hip/hip_bf16.h>

typedef __attribute__((ext_vector_type(16))) _Float16 v16h;
typedef __attribute__((ext_vector_type(2)))  __fp16   v2hp;   // builtin return type
typedef __attribute__((ext_vector_type(8)))  float    v8f;
typedef __attribute__((ext_vector_type(8)))  unsigned int v8u;

#define LAT 128

static __device__ __forceinline__ unsigned int pk2(float a, float b) {
    v2hp p = __builtin_amdgcn_cvt_pkrtz(a, b);   // v_cvt_pk_rtz_f16_f32
    return __builtin_bit_cast(unsigned int, p);
}

static __device__ __forceinline__ v8f wmma_f16(v16h a, v16h b, v8f c) {
    // D = A(16x32 f16) * B(32x16 f16) + C(16x16 f32)
    return __builtin_amdgcn_wmma_f32_16x16x32_f16(false, a, false, b, (short)0, c, false, false);
}

// Relaxed agent-scope fp32 add -> hardware global_atomic_add_f32 (no-return)
static __device__ __forceinline__ void atomic_add_f32(float* p, float v) {
    (void)__hip_atomic_fetch_add(p, v, __ATOMIC_RELAXED, __HIP_MEMORY_SCOPE_AGENT);
}

// Load one 16x32-f16 fragment stored as u32[j][lane] (conflict-free ds loads)
static __device__ __forceinline__ v16h ld_frag(const unsigned int* p, int lane) {
    v8u u;
#pragma unroll
    for (int j = 0; j < 8; ++j) u[j] = p[j * 32 + lane];
    return __builtin_bit_cast(v16h, u);
}

__global__ void gnb_zero_kernel(float4* __restrict__ p, long n4) {
    long i = (long)blockIdx.x * blockDim.x + threadIdx.x;
    long stride = (long)gridDim.x * blockDim.x;
    float4 z = make_float4(0.f, 0.f, 0.f, 0.f);
    for (; i < n4; i += stride) p[i] = z;
}

// ---------------------------------------------------------------------------
// Edge kernel: new_e = LN(relu(relu([ns|nr|e] W1 + b1) W2 + b2)); out = new_e + e
//              aggr[recv] += new_e  (atomic scatter, pre-residual value)
// One wave handles a 16-edge tile x 128 output columns.
// ---------------------------------------------------------------------------
__global__ void __launch_bounds__(256) gnb_edge_kernel(
    const float* __restrict__ nodeL, const float* __restrict__ edgeL,
    const int* __restrict__ senders, const int* __restrict__ receivers,
    const float* __restrict__ w1, const float* __restrict__ b1,
    const float* __restrict__ w2, const float* __restrict__ b2,
    const float* __restrict__ gam, const float* __restrict__ bet,
    float* __restrict__ outE, float* __restrict__ aggr, int nTiles)
{
    extern __shared__ char smem[];
    unsigned int* w1F = (unsigned int*)smem;   // 12 K-blocks * 8 Ntiles * 8 j * 32 lanes
    unsigned int* w2F = w1F + 12 * 2048;       //  4 K-blocks * 2048
    _Float16* h1base  = (_Float16*)(w1F + 16 * 2048);  // per-wave 4*8*32 u32 = 4KB

    const int tid  = threadIdx.x;
    const int lane = tid & 31;
    const int wave = tid >> 5;

    // Convert W1/W2 f32 -> f16 B-fragments in LDS (one-time per block).
    for (int s = tid; s < 16 * 2048; s += 256) {
        const float* W; int rem;
        if (s < 12 * 2048) { W = w1; rem = s; } else { W = w2; rem = s - 12 * 2048; }
        int l  = rem & 31;
        int j  = (rem >> 5) & 7;
        int t  = (rem >> 8) & 7;
        int kb = rem >> 11;
        int k  = kb * 32 + ((l >= 16) ? 8 : 0) + 2 * j + ((j >= 4) ? 8 : 0);   // B layout: lanes>=16 hold K=8..15,24..31
        int n  = t * 16 + (l & 15);
        w1F[s] = pk2(W[k * LAT + n], W[(k + 1) * LAT + n]); // w1F is base of whole area
    }
    __syncthreads();

    float b1v[8], b2v[8], gv[8], bev[8];
#pragma unroll
    for (int t = 0; t < 8; ++t) {
        int n = t * 16 + (lane & 15);
        b1v[t] = b1[n]; b2v[t] = b2[n]; gv[t] = gam[n]; bev[t] = bet[n];
    }

    _Float16* myH1 = h1base + wave * (4 * 8 * 32 * 2);
    const int m   = lane & 15;
    const int hi8 = (lane >= 16) ? 8 : 0;   // A layout K-offset for upper half-wave
    const int m0  = (lane >= 16) ? 8 : 0;   // C/D layout row base for upper half-wave

    for (int tile = blockIdx.x * 8 + wave; tile < nTiles; tile += gridDim.x * 8) {
        const int row  = tile * 16 + m;
        const int srow = senders[row];
        const int rrow = receivers[row];

        // ---- GEMM1: [16x384] x [384x128], K in 12 blocks of 32 ----
        v8f acc[8] = {};
#pragma unroll
        for (int kb = 0; kb < 12; ++kb) {
            const float* src; int off;
            if (kb < 4)      { src = nodeL + (size_t)srow * LAT; off = kb * 32; }
            else if (kb < 8) { src = nodeL + (size_t)rrow * LAT; off = kb * 32 - 128; }
            else             { src = edgeL + (size_t)row  * LAT; off = kb * 32 - 256; }
            float4 fa = *(const float4*)(src + off + hi8);
            float4 fb = *(const float4*)(src + off + hi8 + 4);
            float4 fc = *(const float4*)(src + off + hi8 + 16);
            float4 fd = *(const float4*)(src + off + hi8 + 20);
            v8u au;
            au[0] = pk2(fa.x, fa.y); au[1] = pk2(fa.z, fa.w);
            au[2] = pk2(fb.x, fb.y); au[3] = pk2(fb.z, fb.w);
            au[4] = pk2(fc.x, fc.y); au[5] = pk2(fc.z, fc.w);
            au[6] = pk2(fd.x, fd.y); au[7] = pk2(fd.z, fd.w);
            v16h A = __builtin_bit_cast(v16h, au);
#pragma unroll
            for (int t = 0; t < 8; ++t) {
                v16h B = ld_frag(w1F + (kb * 8 + t) * 256, lane);
                acc[t] = wmma_f16(A, B, acc[t]);
            }
        }

        // ---- bias+relu, transpose h1 into A-fragment layout in LDS ----
#pragma unroll
        for (int t = 0; t < 8; ++t) {
            int n   = t * 16 + (lane & 15);
            int kk  = n & 31, kb2 = n >> 5;
            int q   = kk >> 3;
            int j   = ((q >> 1) ? 4 : 0) + ((kk & 7) >> 1);
            int flA = (q & 1) ? 16 : 0;
            int sel = kk & 1;
#pragma unroll
            for (int r = 0; r < 8; ++r) {
                float h = acc[t][r] + b1v[t];
                h = h > 0.f ? h : 0.f;
                int fl = m0 + r + flA;
                myH1[((kb2 * 8 + j) * 32 + fl) * 2 + sel] = (_Float16)h;
            }
        }
        __builtin_amdgcn_wave_barrier();   // DS ops are in-order per wave

        // ---- GEMM2: [16x128] x [128x128], K in 4 blocks of 32 ----
        v8f acc2[8] = {};
#pragma unroll
        for (int kb = 0; kb < 4; ++kb) {
            v16h A = ld_frag((const unsigned int*)myH1 + kb * 256, lane);
#pragma unroll
            for (int t = 0; t < 8; ++t) {
                v16h B = ld_frag(w2F + (kb * 8 + t) * 256, lane);
                acc2[t] = wmma_f16(A, B, acc2[t]);
            }
        }

        // ---- bias + relu + LayerNorm (per-row over 128 cols) ----
        float ls[8] = {}, lss[8] = {};
#pragma unroll
        for (int t = 0; t < 8; ++t)
#pragma unroll
            for (int r = 0; r < 8; ++r) {
                float h = acc2[t][r] + b2v[t];
                h = h > 0.f ? h : 0.f;
                acc2[t][r] = h;
                ls[r] += h; lss[r] += h * h;
            }
        float mu[8], rstd[8];
#pragma unroll
        for (int r = 0; r < 8; ++r) {
            float s = ls[r], ss = lss[r];
#pragma unroll
            for (int d = 1; d < 16; d <<= 1) {    // reduce within 16-lane half
                s  += __shfl_xor(s, d, 32);
                ss += __shfl_xor(ss, d, 32);
            }
            float mean = s * (1.f / 128.f);
            float var  = ss * (1.f / 128.f) - mean * mean;
            mu[r]   = mean;
            rstd[r] = rsqrtf(var + 1e-5f);
        }

        int rcv8[8];
#pragma unroll
        for (int r = 0; r < 8; ++r) rcv8[r] = receivers[tile * 16 + m0 + r];

        // ---- write residual output + atomic scatter-add to aggr ----
#pragma unroll
        for (int t = 0; t < 8; ++t) {
            int n = t * 16 + (lane & 15);
#pragma unroll
            for (int r = 0; r < 8; ++r) {
                int erow = tile * 16 + m0 + r;
                float v = (acc2[t][r] - mu[r]) * rstd[r] * gv[t] + bev[t];
                outE[(size_t)erow * LAT + n] = v + edgeL[(size_t)erow * LAT + n];
                atomic_add_f32(&aggr[(size_t)rcv8[r] * LAT + n], v);
            }
        }
    }
}

// ---------------------------------------------------------------------------
// Node kernel: out_n = LN(relu(relu([n|aggr] W1 + b1) W2 + b2)) + n
// ---------------------------------------------------------------------------
__global__ void __launch_bounds__(256) gnb_node_kernel(
    const float* __restrict__ nodeL, const float* __restrict__ aggr,
    const float* __restrict__ w1, const float* __restrict__ b1,
    const float* __restrict__ w2, const float* __restrict__ b2,
    const float* __restrict__ gam, const float* __restrict__ bet,
    float* __restrict__ outN, int nTiles)
{
    extern __shared__ char smem[];
    unsigned int* w1F = (unsigned int*)smem;   // 8 K-blocks * 2048
    unsigned int* w2F = w1F + 8 * 2048;        // 4 K-blocks * 2048
    _Float16* h1base  = (_Float16*)(w1F + 12 * 2048);

    const int tid  = threadIdx.x;
    const int lane = tid & 31;
    const int wave = tid >> 5;

    for (int s = tid; s < 12 * 2048; s += 256) {
        const float* W; int rem;
        if (s < 8 * 2048) { W = w1; rem = s; } else { W = w2; rem = s - 8 * 2048; }
        int l  = rem & 31;
        int j  = (rem >> 5) & 7;
        int t  = (rem >> 8) & 7;
        int kb = rem >> 11;
        int k  = kb * 32 + ((l >= 16) ? 8 : 0) + 2 * j + ((j >= 4) ? 8 : 0);
        int n  = t * 16 + (l & 15);
        w1F[s] = pk2(W[k * LAT + n], W[(k + 1) * LAT + n]);
    }
    __syncthreads();

    float b1v[8], b2v[8], gv[8], bev[8];
#pragma unroll
    for (int t = 0; t < 8; ++t) {
        int n = t * 16 + (lane & 15);
        b1v[t] = b1[n]; b2v[t] = b2[n]; gv[t] = gam[n]; bev[t] = bet[n];
    }

    _Float16* myH1 = h1base + wave * (4 * 8 * 32 * 2);
    const int m   = lane & 15;
    const int hi8 = (lane >= 16) ? 8 : 0;
    const int m0  = (lane >= 16) ? 8 : 0;

    for (int tile = blockIdx.x * 8 + wave; tile < nTiles; tile += gridDim.x * 8) {
        const int row = tile * 16 + m;

        v8f acc[8] = {};
#pragma unroll
        for (int kb = 0; kb < 8; ++kb) {
            const float* src = (kb < 4) ? (nodeL + (size_t)row * LAT + kb * 32)
                                        : (aggr  + (size_t)row * LAT + (kb - 4) * 32);
            float4 fa = *(const float4*)(src + hi8);
            float4 fb = *(const float4*)(src + hi8 + 4);
            float4 fc = *(const float4*)(src + hi8 + 16);
            float4 fd = *(const float4*)(src + hi8 + 20);
            v8u au;
            au[0] = pk2(fa.x, fa.y); au[1] = pk2(fa.z, fa.w);
            au[2] = pk2(fb.x, fb.y); au[3] = pk2(fb.z, fb.w);
            au[4] = pk2(fc.x, fc.y); au[5] = pk2(fc.z, fc.w);
            au[6] = pk2(fd.x, fd.y); au[7] = pk2(fd.z, fd.w);
            v16h A = __builtin_bit_cast(v16h, au);
#pragma unroll
            for (int t = 0; t < 8; ++t) {
                v16h B = ld_frag(w1F + (kb * 8 + t) * 256, lane);
                acc[t] = wmma_f16(A, B, acc[t]);
            }
        }

#pragma unroll
        for (int t = 0; t < 8; ++t) {
            int n   = t * 16 + (lane & 15);
            int kk  = n & 31, kb2 = n >> 5;
            int q   = kk >> 3;
            int j   = ((q >> 1) ? 4 : 0) + ((kk & 7) >> 1);
            int flA = (q & 1) ? 16 : 0;
            int sel = kk & 1;
#pragma unroll
            for (int r = 0; r < 8; ++r) {
                float h = acc[t][r] + b1v[t];
                h = h > 0.f ? h : 0.f;
                int fl = m0 + r + flA;
                myH1[((kb2 * 8 + j) * 32 + fl) * 2 + sel] = (_Float16)h;
            }
        }
        __builtin_amdgcn_wave_barrier();

        v8f acc2[8] = {};
#pragma unroll
        for (int kb = 0; kb < 4; ++kb) {
            v16h A = ld_frag((const unsigned int*)myH1 + kb * 256, lane);
#pragma unroll
            for (int t = 0; t < 8; ++t) {
                v16h B = ld_frag(w2F + (kb * 8 + t) * 256, lane);
                acc2[t] = wmma_f16(A, B, acc2[t]);
            }
        }

        float ls[8] = {}, lss[8] = {};
#pragma unroll
        for (int t = 0; t < 8; ++t)
#pragma unroll
            for (int r = 0; r < 8; ++r) {
                float h = acc2[t][r] + b2v[t];
                h = h > 0.f ? h : 0.f;
                acc2[t][r] = h;
                ls[r] += h; lss[r] += h * h;
            }
        float mu[8], rstd[8];
#pragma unroll
        for (int r = 0; r < 8; ++r) {
            float s = ls[r], ss = lss[r];
#pragma unroll
            for (int d = 1; d < 16; d <<= 1) {
                s  += __shfl_xor(s, d, 32);
                ss += __shfl_xor(ss, d, 32);
            }
            float mean = s * (1.f / 128.f);
            float var  = ss * (1.f / 128.f) - mean * mean;
            mu[r]   = mean;
            rstd[r] = rsqrtf(var + 1e-5f);
        }

#pragma unroll
        for (int t = 0; t < 8; ++t) {
            int n = t * 16 + (lane & 15);
#pragma unroll
            for (int r = 0; r < 8; ++r) {
                int nrow = tile * 16 + m0 + r;
                float v = (acc2[t][r] - mu[r]) * rstd[r] * gv[t] + bev[t];
                outN[(size_t)nrow * LAT + n] = v + nodeL[(size_t)nrow * LAT + n];
            }
        }
    }
}

extern "C" void kernel_launch(void* const* d_in, const int* in_sizes, int n_in,
                              void* d_out, int out_size, void* d_ws, size_t ws_size,
                              hipStream_t stream) {
    const float* nodeL = (const float*)d_in[0];
    const float* edgeL = (const float*)d_in[1];
    const int*   snd   = (const int*)d_in[2];
    const int*   rcv   = (const int*)d_in[3];
    const float* me_w1 = (const float*)d_in[4];
    const float* me_b1 = (const float*)d_in[5];
    const float* me_w2 = (const float*)d_in[6];
    const float* me_b2 = (const float*)d_in[7];
    const float* me_g  = (const float*)d_in[8];
    const float* me_be = (const float*)d_in[9];
    const float* nf_w1 = (const float*)d_in[10];
    const float* nf_b1 = (const float*)d_in[11];
    const float* nf_w2 = (const float*)d_in[12];
    const float* nf_b2 = (const float*)d_in[13];
    const float* nf_g  = (const float*)d_in[14];
    const float* nf_be = (const float*)d_in[15];

    const int nNodes = in_sizes[0] / LAT;
    const int nEdges = in_sizes[1] / LAT;

    float* outN = (float*)d_out;
    float* outE = outN + (size_t)nNodes * LAT;
    float* aggr = (float*)d_ws;   // nNodes * 128 f32 scratch

    {   // zero aggregation buffer every call (deterministic)
        long n4 = (long)nNodes * (LAT / 4);
        int grid = (int)((n4 + 255) / 256);
        if (grid > 4096) grid = 4096;
        gnb_zero_kernel<<<grid, 256, 0, stream>>>((float4*)aggr, n4);
    }
    {   // edge MLP + LN + residual + scatter-add
        int tiles = nEdges / 16;
        int grid = 960;
        if (grid * 8 > tiles) grid = (tiles + 7) / 8;
        size_t lds = (size_t)16 * 2048 * 4 + 8 * 4096;   // 163840 B
        gnb_edge_kernel<<<grid, 256, lds, stream>>>(nodeL, edgeL, snd, rcv,
            me_w1, me_b1, me_w2, me_b2, me_g, me_be, outE, aggr, tiles);
    }
    {   // node MLP + LN + residual
        int tiles = nNodes / 16;
        int grid = (tiles + 7) / 8;
        if (grid > 512) grid = 512;
        size_t lds = (size_t)12 * 2048 * 4 + 8 * 4096;   // 131072 B
        gnb_node_kernel<<<grid, 256, lds, stream>>>(nodeL, aggr,
            nf_w1, nf_b1, nf_w2, nf_b2, nf_g, nf_be, outN, tiles);
    }
}
